// Block_21947282883003
// MI455X (gfx1250) — matrix operand order repaired and checked by reading, not compiled, not run on hardware
//
#include <hip/hip_runtime.h>
#include <hip/hip_bf16.h>
#include <math.h>

// ---------------------------------------------------------------------------
// Types for CDNA5 WMMA (wave32)
// ---------------------------------------------------------------------------
typedef __bf16 bf16;
typedef __attribute__((ext_vector_type(16))) __bf16 v16bf;
typedef __attribute__((ext_vector_type(8)))  __bf16 v8bf;
typedef __attribute__((ext_vector_type(8)))  float  v8f;

#define WMMA_BF16(a, b, c) \
  __builtin_amdgcn_wmma_f32_16x16x32_bf16(false, (a), false, (b), (short)0, (c), false, false)

// Model constants
#define DMODEL 768
#define NHEAD  12
#define HS     64
#define SEQ    2048
#define NB     2
#define MTOT   (NB * SEQ)   // 4096 rows
#define DFF    (4 * DMODEL) // 3072

// ---------------------------------------------------------------------------
// Fragment loaders from LDS (layouts per CDNA5 ISA §7.12.2)
//  A 16x32 bf16: lane half hl, element e: k = (e<8 ? hl*8+e : 16+hl*8+(e-8))
//  B 32x16 bf16 (read as Bt[n][k]): element e: k = (e<8 ? hl*8+e : 16+hl*8+(e-8))
// ---------------------------------------------------------------------------
__device__ __forceinline__ v16bf frag_a(const bf16* rowbase, int hl) {
  v8bf lo = *(const v8bf*)(rowbase + hl * 8);
  v8bf hi = *(const v8bf*)(rowbase + 16 + hl * 8);
  v16bf f;
#pragma unroll
  for (int i = 0; i < 8; ++i) { f[i] = lo[i]; f[i + 8] = hi[i]; }
  return f;
}

__device__ __forceinline__ v16bf frag_b(const bf16* rowbase, int hl) {
  v8bf lo = *(const v8bf*)(rowbase + hl * 8);
  v8bf hi = *(const v8bf*)(rowbase + 16 + hl * 8);
  v16bf f;
#pragma unroll
  for (int i = 0; i < 8; ++i) { f[i] = lo[i]; f[i + 8] = hi[i]; }
  return f;
}

// ---------------------------------------------------------------------------
// Shared 128x128-block GEMM mainloop with register double-buffering:
// prefetch k-tile (kt+32) from global into VGPRs while tile kt computes.
// 256 threads = 8 waves; wave (wm in 0..1, wn in 0..3) owns a 64x32 tile.
// ---------------------------------------------------------------------------
__device__ __forceinline__ void gemm_mainloop(const bf16* __restrict__ A,
                                              const bf16* __restrict__ Bt,
                                              int K, int m0, int n0,
                                              bf16* As, bf16* Bs,
                                              v8f acc[4][2]) {
  const int tid  = threadIdx.x;
  const int lane = tid & 31, hl = lane >> 4, ln = lane & 15;
  const int wid  = tid >> 5, wm = wid & 1, wn = wid >> 1;
  const int r    = tid >> 1, kh = (tid & 1) << 4;

  const bf16* pa = A + (size_t)(m0 + r) * K + kh;
  const bf16* pb = Bt + (size_t)(n0 + r) * K + kh;

  // Prologue: fetch k-tile 0 into registers
  v8bf ra0 = *(const v8bf*)pa;
  v8bf ra1 = *(const v8bf*)(pa + 8);
  v8bf rb0 = *(const v8bf*)pb;
  v8bf rb1 = *(const v8bf*)(pb + 8);

  for (int kt = 0; kt < K; kt += 32) {
    __syncthreads();  // previous tile's LDS reads done
    *(v8bf*)&As[r * 40 + kh]     = ra0;
    *(v8bf*)&As[r * 40 + kh + 8] = ra1;
    *(v8bf*)&Bs[r * 40 + kh]     = rb0;
    *(v8bf*)&Bs[r * 40 + kh + 8] = rb1;
    __syncthreads();

    // Prefetch next k-tile: overlaps the ds_loads + 8 WMMAs below
    if (kt + 32 < K) {
      ra0 = *(const v8bf*)(pa + kt + 32);
      ra1 = *(const v8bf*)(pa + kt + 40);
      rb0 = *(const v8bf*)(pb + kt + 32);
      rb1 = *(const v8bf*)(pb + kt + 40);
    }

    v16bf a[4], b[2];
#pragma unroll
    for (int mt = 0; mt < 4; ++mt)
      a[mt] = frag_a(&As[(wm * 64 + mt * 16 + ln) * 40], hl);
#pragma unroll
    for (int nt = 0; nt < 2; ++nt)
      b[nt] = frag_b(&Bs[(wn * 32 + nt * 16 + ln) * 40], hl);
#pragma unroll
    for (int mt = 0; mt < 4; ++mt)
#pragma unroll
      for (int nt = 0; nt < 2; ++nt)
        acc[mt][nt] = WMMA_BF16(a[mt], b[nt], acc[mt][nt]);
  }
}

// ---------------------------------------------------------------------------
// Generic GEMM: out = [relu](A @ Bt^T + bias) [+ res], out fp32 or bf16
// ---------------------------------------------------------------------------
template <int OUT_BF16, int RELU, int HAS_RES>
__global__ __launch_bounds__(256) void gemm_bt_kernel(
    const bf16* __restrict__ A, const bf16* __restrict__ Bt,
    const float* __restrict__ bias, const float* __restrict__ res,
    float* __restrict__ outf, bf16* __restrict__ outb, int N, int K) {
  __shared__ alignas(16) bf16 As[128 * 40];
  __shared__ alignas(16) bf16 Bs[128 * 40];
  const int m0 = blockIdx.y * 128, n0 = blockIdx.x * 128;
  v8f acc[4][2] = {};
  gemm_mainloop(A, Bt, K, m0, n0, As, Bs, acc);

  const int tid = threadIdx.x, lane = tid & 31, hl = lane >> 4, ln = lane & 15;
  const int wid = tid >> 5, wm = wid & 1, wn = wid >> 1;
#pragma unroll
  for (int mt = 0; mt < 4; ++mt)
#pragma unroll
    for (int nt = 0; nt < 2; ++nt) {
      const int col = n0 + wn * 32 + nt * 16 + ln;
      const float bv = bias[col];
#pragma unroll
      for (int e = 0; e < 8; ++e) {
        const int row = m0 + wm * 64 + mt * 16 + e + 8 * hl;
        float v = acc[mt][nt][e] + bv;
        if (RELU) v = fmaxf(v, 0.f);
        if (HAS_RES) v += res[(size_t)row * N + col];
        if (OUT_BF16) outb[(size_t)row * N + col] = (bf16)v;
        else          outf[(size_t)row * N + col] = v;
      }
    }
}

// ---------------------------------------------------------------------------
// Fused QKV GEMM (blockIdx.z selects Wq/Wk/Wv). Q,K written per-head
// [b,h,t,hs]; V written per-head TRANSPOSED [b,h,hs,t] for attention P@V.
// ---------------------------------------------------------------------------
__global__ __launch_bounds__(256) void gemm_qkv_kernel(
    const bf16* __restrict__ A, const bf16* __restrict__ Wqt,
    const bf16* __restrict__ Wkt, const bf16* __restrict__ Wvt,
    bf16* __restrict__ Qg, bf16* __restrict__ Kg, bf16* __restrict__ Vtg) {
  __shared__ alignas(16) bf16 As[128 * 40];
  __shared__ alignas(16) bf16 Bs[128 * 40];
  const int z = blockIdx.z;
  const bf16* Bt = (z == 0) ? Wqt : (z == 1) ? Wkt : Wvt;
  const int m0 = blockIdx.y * 128, n0 = blockIdx.x * 128;
  v8f acc[4][2] = {};
  gemm_mainloop(A, Bt, DMODEL, m0, n0, As, Bs, acc);

  const int tid = threadIdx.x, lane = tid & 31, hl = lane >> 4, ln = lane & 15;
  const int wid = tid >> 5, wm = wid & 1, wn = wid >> 1;
#pragma unroll
  for (int mt = 0; mt < 4; ++mt)
#pragma unroll
    for (int nt = 0; nt < 2; ++nt) {
      const int col = n0 + wn * 32 + nt * 16 + ln;
      const int hh = col >> 6, kk = col & (HS - 1);
#pragma unroll
      for (int e = 0; e < 8; ++e) {
        const int row = m0 + wm * 64 + mt * 16 + e + 8 * hl;
        const int bb = row >> 11, t = row & (SEQ - 1);
        const bf16 v = (bf16)acc[mt][nt][e];
        if (z == 0)
          Qg[(((size_t)(bb * NHEAD + hh)) * SEQ + t) * HS + kk] = v;
        else if (z == 1)
          Kg[(((size_t)(bb * NHEAD + hh)) * SEQ + t) * HS + kk] = v;
        else
          Vtg[(((size_t)(bb * NHEAD + hh)) * HS + kk) * SEQ + t] = v;
      }
    }
}

// ---------------------------------------------------------------------------
// LayerNorm: one block per row of 768; out = bf16 normalized row
// ---------------------------------------------------------------------------
__global__ __launch_bounds__(256) void ln_kernel(const float* __restrict__ x,
                                                 const float* __restrict__ g,
                                                 const float* __restrict__ b,
                                                 bf16* __restrict__ out) {
  __shared__ float sred[8];
  const int row = blockIdx.x, tid = threadIdx.x;
  const float* xr = x + (size_t)row * DMODEL;
  float v0 = xr[tid], v1 = xr[tid + 256], v2 = xr[tid + 512];
  float s = v0 + v1 + v2;
#pragma unroll
  for (int m = 16; m >= 1; m >>= 1) s += __shfl_xor(s, m, 32);
  if ((tid & 31) == 0) sred[tid >> 5] = s;
  __syncthreads();
  float tot = 0.f;
#pragma unroll
  for (int i = 0; i < 8; ++i) tot += sred[i];
  const float mean = tot * (1.f / DMODEL);
  float d0 = v0 - mean, d1 = v1 - mean, d2 = v2 - mean;
  float q = d0 * d0 + d1 * d1 + d2 * d2;
  __syncthreads();
#pragma unroll
  for (int m = 16; m >= 1; m >>= 1) q += __shfl_xor(q, m, 32);
  if ((tid & 31) == 0) sred[tid >> 5] = q;
  __syncthreads();
  float qt = 0.f;
#pragma unroll
  for (int i = 0; i < 8; ++i) qt += sred[i];
  const float inv = rsqrtf(qt * (1.f / DMODEL) + 1e-5f);
  bf16* orow = out + (size_t)row * DMODEL;
  orow[tid]       = (bf16)(d0 * inv * g[tid]       + b[tid]);
  orow[tid + 256] = (bf16)(d1 * inv * g[tid + 256] + b[tid + 256]);
  orow[tid + 512] = (bf16)(d2 * inv * g[tid + 512] + b[tid + 512]);
}

// ---------------------------------------------------------------------------
// Weight convert + transpose: W[k][n] fp32 -> Wt[n][k] bf16
// ---------------------------------------------------------------------------
__global__ __launch_bounds__(256) void wt_kernel(const float* __restrict__ W,
                                                 bf16* __restrict__ Wt,
                                                 int K, int N) {
  const int n = blockIdx.x * 32 + (threadIdx.x & 31);
  const int k = blockIdx.y * 8 + (threadIdx.x >> 5);
  Wt[(size_t)n * K + k] = (bf16)W[(size_t)k * N + n];
}

// ---------------------------------------------------------------------------
// Flash attention: block = (b, h, 128 query rows); 8 waves x 16 rows.
// Streams 64-key tiles with register double-buffered K/V staging;
// online softmax; all matmuls via WMMA bf16.
// ---------------------------------------------------------------------------
__global__ __launch_bounds__(256) void attn_kernel(
    const bf16* __restrict__ Qg, const bf16* __restrict__ Kg,
    const bf16* __restrict__ Vtg, const int* __restrict__ mask,
    bf16* __restrict__ Og) {
  __shared__ alignas(16) bf16 Qs[128 * 72];
  __shared__ alignas(16) bf16 Ks[64 * 72];
  __shared__ alignas(16) bf16 Vs[64 * 72];
  __shared__ alignas(16) bf16 Ps[8 * 16 * 72];
  __shared__ float madd[64];

  const int bb = blockIdx.z, hh = blockIdx.y, m0 = blockIdx.x * 128;
  const int tid = threadIdx.x, lane = tid & 31, wid = tid >> 5;
  const int hl = lane >> 4, ln = lane & 15;

  const bf16* Qh = Qg + ((size_t)(bb * NHEAD + hh) * SEQ) * HS;
  const bf16* Kh = Kg + ((size_t)(bb * NHEAD + hh) * SEQ) * HS;
  const bf16* Vh = Vtg + ((size_t)(bb * NHEAD + hh) * HS) * SEQ;

  const int rr = tid >> 2, seg = (tid & 3) << 4;

  // Stage Q tile (128 x 64)
#pragma unroll
  for (int it = 0; it < 2; ++it) {
    const int r = it * 64 + rr;
    const bf16* src = Qh + (size_t)(m0 + r) * HS + seg;
    *(v8bf*)&Qs[r * 72 + seg]     = *(const v8bf*)src;
    *(v8bf*)&Qs[r * 72 + seg + 8] = *(const v8bf*)(src + 8);
  }
  __syncthreads();

  v16bf qa[2];
  qa[0] = frag_a(&Qs[(wid * 16 + ln) * 72], hl);
  qa[1] = frag_a(&Qs[(wid * 16 + ln) * 72 + 32], hl);

  float mrow[8], lrow[8];
  v8f o[4] = {};
#pragma unroll
  for (int r = 0; r < 8; ++r) { mrow[r] = -INFINITY; lrow[r] = 0.f; }
  const float scale = rsqrtf((float)DMODEL);  // NOTE: embed-dim scaling
  bf16* Pw = &Ps[wid * 16 * 72];

  // Register double-buffer: prologue fetch of s-tile 0
  const bf16* kp = Kh + (size_t)rr * HS + seg;  // advance by s0*HS per tile
  const bf16* vp = Vh + (size_t)rr * SEQ + seg; // advance by s0 per tile
  v8bf rk0 = *(const v8bf*)kp;
  v8bf rk1 = *(const v8bf*)(kp + 8);
  v8bf rv0 = *(const v8bf*)vp;
  v8bf rv1 = *(const v8bf*)(vp + 8);
  float mreg = (tid < 64) ? ((mask[bb * SEQ + tid] == 0) ? -INFINITY : 0.f) : 0.f;

  for (int s0 = 0; s0 < SEQ; s0 += 64) {
    __syncthreads();  // previous tile's LDS reads done
    *(v8bf*)&Ks[rr * 72 + seg]     = rk0;
    *(v8bf*)&Ks[rr * 72 + seg + 8] = rk1;
    *(v8bf*)&Vs[rr * 72 + seg]     = rv0;
    *(v8bf*)&Vs[rr * 72 + seg + 8] = rv1;
    if (tid < 64) madd[tid] = mreg;
    __syncthreads();

    // Prefetch next s-tile: overlaps the 16 WMMAs + softmax below
    const int sn = s0 + 64;
    if (sn < SEQ) {
      rk0 = *(const v8bf*)(kp + (size_t)sn * HS);
      rk1 = *(const v8bf*)(kp + (size_t)sn * HS + 8);
      rv0 = *(const v8bf*)(vp + sn);
      rv1 = *(const v8bf*)(vp + sn + 8);
      if (tid < 64) mreg = (mask[bb * SEQ + sn + tid] == 0) ? -INFINITY : 0.f;
    }

    // S = Q * K^T (16x64 per wave)
    v8f sac[4] = {};
#pragma unroll
    for (int ks = 0; ks < 2; ++ks)
#pragma unroll
      for (int nt = 0; nt < 4; ++nt) {
        v16bf bbf = frag_b(&Ks[(nt * 16 + ln) * 72 + ks * 32], hl);
        sac[nt] = WMMA_BF16(qa[ks], bbf, sac[nt]);
      }

    // Online softmax (rows r+8*hl live across the 16 lanes of half hl)
    float pm[8];
#pragma unroll
    for (int r = 0; r < 8; ++r) pm[r] = -INFINITY;
#pragma unroll
    for (int nt = 0; nt < 4; ++nt) {
      const float ad = madd[nt * 16 + ln];
#pragma unroll
      for (int r = 0; r < 8; ++r) {
        float sv = sac[nt][r] * scale + ad;
        sac[nt][r] = sv;
        pm[r] = fmaxf(pm[r], sv);
      }
    }
#pragma unroll
    for (int m = 8; m >= 1; m >>= 1)
#pragma unroll
      for (int r = 0; r < 8; ++r) pm[r] = fmaxf(pm[r], __shfl_xor(pm[r], m, 32));

    float alpha[8], rs[8];
#pragma unroll
    for (int r = 0; r < 8; ++r) {
      const float mn = fmaxf(mrow[r], pm[r]);
      alpha[r] = (mrow[r] == mn) ? 1.f : __expf(mrow[r] - mn);
      mrow[r] = mn;
      rs[r] = 0.f;
    }
#pragma unroll
    for (int nt = 0; nt < 4; ++nt)
#pragma unroll
      for (int r = 0; r < 8; ++r) {
        const float p = __expf(sac[nt][r] - mrow[r]);
        rs[r] += p;
        Pw[(r + 8 * hl) * 72 + nt * 16 + ln] = (bf16)p;
      }
#pragma unroll
    for (int m = 8; m >= 1; m >>= 1)
#pragma unroll
      for (int r = 0; r < 8; ++r) rs[r] += __shfl_xor(rs[r], m, 32);
#pragma unroll
    for (int r = 0; r < 8; ++r) lrow[r] = lrow[r] * alpha[r] + rs[r];
#pragma unroll
    for (int nt = 0; nt < 4; ++nt)
#pragma unroll
      for (int r = 0; r < 8; ++r) o[nt][r] *= alpha[r];

    // O += P * V (P re-read from LDS in A-fragment layout; same-wave LDS in order)
#pragma unroll
    for (int ks = 0; ks < 2; ++ks) {
      v16bf pa = frag_a(&Pw[ln * 72 + ks * 32], hl);
#pragma unroll
      for (int nt = 0; nt < 4; ++nt) {
        v16bf vb = frag_b(&Vs[(nt * 16 + ln) * 72 + ks * 32], hl);
        o[nt] = WMMA_BF16(pa, vb, o[nt]);
      }
    }
  }

  // Normalize and write concat-head output [B*T, D] bf16
#pragma unroll
  for (int nt = 0; nt < 4; ++nt)
#pragma unroll
    for (int r = 0; r < 8; ++r) {
      const int row = m0 + wid * 16 + r + 8 * hl;
      const int col = hh * HS + nt * 16 + ln;
      Og[(size_t)(bb * SEQ + row) * DMODEL + col] = (bf16)(o[nt][r] / lrow[r]);
    }
}

// ---------------------------------------------------------------------------
// Host launcher
// ---------------------------------------------------------------------------
extern "C" void kernel_launch(void* const* d_in, const int* in_sizes, int n_in,
                              void* d_out, int out_size, void* d_ws, size_t ws_size,
                              hipStream_t stream) {
  (void)in_sizes; (void)n_in; (void)out_size; (void)ws_size;
  const float* x    = (const float*)d_in[0];
  const int*  amsk  = (const int*)d_in[1];
  const float* Wq   = (const float*)d_in[2];
  const float* Wk   = (const float*)d_in[3];
  const float* Wv   = (const float*)d_in[4];
  const float* Wo   = (const float*)d_in[5];
  const float* bo   = (const float*)d_in[6];
  const float* ln1g = (const float*)d_in[7];
  const float* ln1b = (const float*)d_in[8];
  const float* ln2g = (const float*)d_in[9];
  const float* ln2b = (const float*)d_in[10];
  const float* W1   = (const float*)d_in[11];
  const float* b1   = (const float*)d_in[12];
  const float* W2   = (const float*)d_in[13];
  const float* b2   = (const float*)d_in[14];
  float* out = (float*)d_out;

  char* ws = (char*)d_ws;
  size_t off = 0;
  auto alloc = [&](size_t bytes) -> char* {
    char* p = ws + off;
    off += (bytes + 255) & ~(size_t)255;
    return p;
  };

  bf16* hbuf = (bf16*)alloc((size_t)MTOT * DMODEL * 2);
  bf16* Wqt  = (bf16*)alloc((size_t)DMODEL * DMODEL * 2);
  bf16* Wkt  = (bf16*)alloc((size_t)DMODEL * DMODEL * 2);
  bf16* Wvt  = (bf16*)alloc((size_t)DMODEL * DMODEL * 2);
  bf16* Wot  = (bf16*)alloc((size_t)DMODEL * DMODEL * 2);
  bf16* W1t  = (bf16*)alloc((size_t)DMODEL * DFF * 2);
  bf16* W2t  = (bf16*)alloc((size_t)DFF * DMODEL * 2);
  bf16* Qg   = (bf16*)alloc((size_t)MTOT * DMODEL * 2);
  bf16* Kg   = (bf16*)alloc((size_t)MTOT * DMODEL * 2);
  bf16* Vtg  = (bf16*)alloc((size_t)MTOT * DMODEL * 2);
  bf16* attn = (bf16*)alloc((size_t)MTOT * DMODEL * 2);
  float* x1  = (float*)alloc((size_t)MTOT * DMODEL * 4);
  bf16* h2   = (bf16*)alloc((size_t)MTOT * DMODEL * 2);
  bf16* ff1  = (bf16*)alloc((size_t)MTOT * DFF * 2);

  // Weight convert+transpose (bf16)
  wt_kernel<<<dim3(DMODEL / 32, DMODEL / 8), 256, 0, stream>>>(Wq, Wqt, DMODEL, DMODEL);
  wt_kernel<<<dim3(DMODEL / 32, DMODEL / 8), 256, 0, stream>>>(Wk, Wkt, DMODEL, DMODEL);
  wt_kernel<<<dim3(DMODEL / 32, DMODEL / 8), 256, 0, stream>>>(Wv, Wvt, DMODEL, DMODEL);
  wt_kernel<<<dim3(DMODEL / 32, DMODEL / 8), 256, 0, stream>>>(Wo, Wot, DMODEL, DMODEL);
  wt_kernel<<<dim3(DFF / 32, DMODEL / 8), 256, 0, stream>>>(W1, W1t, DMODEL, DFF);
  wt_kernel<<<dim3(DMODEL / 32, DFF / 8), 256, 0, stream>>>(W2, W2t, DFF, DMODEL);

  // LN1 -> h
  ln_kernel<<<MTOT, 256, 0, stream>>>(x, ln1g, ln1b, hbuf);

  // QKV projections (fused over z)
  gemm_qkv_kernel<<<dim3(DMODEL / 128, MTOT / 128, 3), 256, 0, stream>>>(
      hbuf, Wqt, Wkt, Wvt, Qg, Kg, Vtg);

  // Flash attention
  attn_kernel<<<dim3(SEQ / 128, NHEAD, NB), 256, 0, stream>>>(Qg, Kg, Vtg, amsk, attn);

  // x1 = x + attn @ Wo + bo  (fp32)
  gemm_bt_kernel<0, 0, 1><<<dim3(DMODEL / 128, MTOT / 128), 256, 0, stream>>>(
      attn, Wot, bo, x, x1, nullptr, DMODEL, DMODEL);

  // LN2 -> h2
  ln_kernel<<<MTOT, 256, 0, stream>>>(x1, ln2g, ln2b, h2);

  // ff1 = relu(h2 @ W1 + b1)  (bf16)
  gemm_bt_kernel<1, 1, 0><<<dim3(DFF / 128, MTOT / 128), 256, 0, stream>>>(
      h2, W1t, b1, nullptr, nullptr, ff1, DFF, DMODEL);

  // out = x1 + ff1 @ W2 + b2  (fp32)
  gemm_bt_kernel<0, 0, 1><<<dim3(DMODEL / 128, MTOT / 128), 256, 0, stream>>>(
      ff1, W2t, b2, x1, out, nullptr, DMODEL, DFF);
}
